// TLSTMLayer_45226005627004
// MI455X (gfx1250) — compile-verified, matrix operand order checked by reading, not run on hardware
//
#include <hip/hip_runtime.h>

#define T_DIM 1024
#define B_DIM 256
#define D_DIM 128
#define H_DIM 256

typedef __attribute__((ext_vector_type(16))) _Float16 v16h;
typedef __attribute__((ext_vector_type(8)))  _Float16 v8h;
typedef __attribute__((ext_vector_type(8)))  float    v8f;

// ---------------------------------------------------------------------------
// Fragment loader following the CDNA5 16-bit A/B operand layout
// (CDNA5 ISA §7.12.2):
//   lanes 0-15 : row = base+lane,    K = k0 + {0..7, 16..23}
//   lanes 16-31: row = base+lane-16, K = k0+8 + {0..7, 16..23}
// Caller passes row = rowbase + (lane&15) and k = k0 + (lane>>4)*8.
// Two contiguous 16B loads per lane -> global_load_b128 x2.
// ---------------------------------------------------------------------------
__device__ __forceinline__ v16h load_frag_h(const _Float16* __restrict__ base,
                                            int ld, int row, int k) {
  const _Float16* p = base + (size_t)row * ld + k;
  v8h lo = *reinterpret_cast<const v8h*>(p);
  v8h hi = *reinterpret_cast<const v8h*>(p + 16);
  return __builtin_shufflevector(lo, hi, 0, 1, 2, 3, 4, 5, 6, 7,
                                 8, 9, 10, 11, 12, 13, 14, 15);
}

// Fast transcendentals: v_exp_f32 / v_rcp_f32 based, saturate correctly.
__device__ __forceinline__ float fast_rcp(float x) {
  return __builtin_amdgcn_rcpf(x);
}
__device__ __forceinline__ float fast_sigmoid(float x) {
  return fast_rcp(1.0f + __expf(-x));
}
__device__ __forceinline__ float fast_tanh(float x) {
  // tanh(x) = 1 - 2/(exp(2x)+1); exp->inf gives 1, exp->0 gives -1.
  return 1.0f - 2.0f * fast_rcp(__expf(2.0f * x) + 1.0f);
}

#define WMMA_F16(A, B, C) \
  __builtin_amdgcn_wmma_f32_16x16x32_f16(false, (A), false, (B), (short)0, (C), false, false)

// ---------------------------------------------------------------------------
// One T-LSTM timestep. Grid: 256 blocks of 1 wave (32 threads).
// Block bx owns the 16x16 tile at rows m0 = (bx>>4)*16, cols n0 = (bx&15)*16
// of h_new / c_new: 4 gate tiles + 1 Cs tile, A-fragment shared across the
// 4 gate WMMAs; the Cs chain is interleaved with the U chains so each k-step
// issues 5 independent WMMAs (hides WMMA->WMMA hazards).
// ---------------------------------------------------------------------------
__global__ __launch_bounds__(32) void tlstm_step(
    const _Float16* __restrict__ xt,       // [B,D] f16 slice of x at step t
    const float* __restrict__ dts,         // [B,T] elapsed times
    int t,
    const _Float16* __restrict__ Wh,       // [4H,D] f16
    const _Float16* __restrict__ Uh,       // [4H,H] f16
    const _Float16* __restrict__ Wdh,      // [H,H]  f16
    const float* __restrict__ b,           // [4H]
    const float* __restrict__ bd,          // [H]
    const _Float16* __restrict__ hprevh,   // [B,H] f16
    const _Float16* __restrict__ cprevh,   // [B,H] f16
    const float* __restrict__ cprevf,      // [B,H] f32
    _Float16* __restrict__ hnexth,         // [B,H] f16
    _Float16* __restrict__ cnexth,         // [B,H] f16
    float* __restrict__ cnextf,            // [B,H] f32
    float* __restrict__ outs,              // [T,B,H] output base
    float* __restrict__ hfin,              // non-null only at t==T-1
    float* __restrict__ cfin)              // non-null only at t==T-1
{
  const int lane = threadIdx.x;            // wave32
  const int col  = lane & 15;              // N within tile (C/D layout)
  const int koff = (lane >> 4) * 8;        // K-offset for A/B fragments
  const int bx   = blockIdx.x;
  const int n0   = (bx & 15) * 16;
  const int m0   = (bx >> 4) * 16;

  v8f acc_i = {}, acc_f = {}, acc_g = {}, acc_o = {}, acc_s = {};

  // ---- gates += x_t @ W^T   (K = D = 128), 4 independent chains ----
#pragma unroll
  for (int k0 = 0; k0 < D_DIM; k0 += 32) {
    v16h a  = load_frag_h(xt, D_DIM, m0 + col, k0 + koff);
    v16h b0 = load_frag_h(Wh, D_DIM, 0 * H_DIM + n0 + col, k0 + koff);
    v16h b1 = load_frag_h(Wh, D_DIM, 1 * H_DIM + n0 + col, k0 + koff);
    v16h b2 = load_frag_h(Wh, D_DIM, 2 * H_DIM + n0 + col, k0 + koff);
    v16h b3 = load_frag_h(Wh, D_DIM, 3 * H_DIM + n0 + col, k0 + koff);
    acc_i = WMMA_F16(a, b0, acc_i);
    acc_f = WMMA_F16(a, b1, acc_f);
    acc_g = WMMA_F16(a, b2, acc_g);
    acc_o = WMMA_F16(a, b3, acc_o);
  }

  // ---- gates += h_prev @ U^T  and  Cs_pre += c_prev @ Wd^T  (K = H = 256)
  //      merged: 5 independent accumulator chains per k-step ----
#pragma unroll
  for (int k0 = 0; k0 < H_DIM; k0 += 32) {
    v16h ah = load_frag_h(hprevh, H_DIM, m0 + col, k0 + koff);
    v16h ac = load_frag_h(cprevh, H_DIM, m0 + col, k0 + koff);
    v16h b0 = load_frag_h(Uh, H_DIM, 0 * H_DIM + n0 + col, k0 + koff);
    v16h b1 = load_frag_h(Uh, H_DIM, 1 * H_DIM + n0 + col, k0 + koff);
    v16h b2 = load_frag_h(Uh, H_DIM, 2 * H_DIM + n0 + col, k0 + koff);
    v16h b3 = load_frag_h(Uh, H_DIM, 3 * H_DIM + n0 + col, k0 + koff);
    v16h bw = load_frag_h(Wdh, H_DIM, n0 + col, k0 + koff);
    acc_i = WMMA_F16(ah, b0, acc_i);
    acc_f = WMMA_F16(ah, b1, acc_f);
    acc_g = WMMA_F16(ah, b2, acc_g);
    acc_o = WMMA_F16(ah, b3, acc_o);
    acc_s = WMMA_F16(ac, bw, acc_s);
  }

  // ---- elementwise T-LSTM update (fp32) ----
  const float bias_i = b[0 * H_DIM + n0 + col];
  const float bias_f = b[1 * H_DIM + n0 + col];
  const float bias_g = b[2 * H_DIM + n0 + col];
  const float bias_o = b[3 * H_DIM + n0 + col];
  const float bias_d = bd[n0 + col];

  const int rbase = m0 + (lane >> 4) * 8;  // C/D layout: VGPR r -> row rbase+r
#pragma unroll
  for (int r = 0; r < 8; ++r) {
    const int row = rbase + r;
    const float dt  = dts[(size_t)row * T_DIM + t];
    const float gdt = fast_rcp(__logf(2.71828182845904523536f + dt));
    const int idx = row * H_DIM + n0 + col;

    const float cp    = cprevf[idx];
    const float cs    = fast_tanh(acc_s[r] + bias_d);
    const float cstar = cp - cs + cs * gdt;

    const float iv = fast_sigmoid(acc_i[r] + bias_i);
    const float fv = fast_sigmoid(acc_f[r] + bias_f);
    const float gv = fast_tanh(acc_g[r] + bias_g);
    const float ov = fast_sigmoid(acc_o[r] + bias_o);

    const float cn = fv * cstar + iv * gv;
    const float hn = ov * fast_tanh(cn);

    cnextf[idx] = cn;
    cnexth[idx] = (_Float16)cn;
    hnexth[idx] = (_Float16)hn;
    outs[(size_t)t * (B_DIM * H_DIM) + idx] = hn;
    if (hfin) {
      hfin[idx] = hn;
      cfin[idx] = cn;
    }
  }
}

// ---------------------------------------------------------------------------
// Setup: f32 -> f16 conversion and f32 copy
// ---------------------------------------------------------------------------
__global__ void cvt_f32_to_h_kernel(const float* __restrict__ in,
                                    _Float16* __restrict__ out, int n) {
  int i = blockIdx.x * blockDim.x + threadIdx.x;
  if (i < n) out[i] = (_Float16)in[i];
}

__global__ void copy_f32_kernel(const float* __restrict__ in,
                                float* __restrict__ out, int n) {
  int i = blockIdx.x * blockDim.x + threadIdx.x;
  if (i < n) out[i] = in[i];
}

// ---------------------------------------------------------------------------
extern "C" void kernel_launch(void* const* d_in, const int* in_sizes, int n_in,
                              void* d_out, int out_size, void* d_ws, size_t ws_size,
                              hipStream_t stream) {
  const float* x   = (const float*)d_in[0];  // [T,B,D]
  const float* dts = (const float*)d_in[1];  // [B,T]
  const float* h0  = (const float*)d_in[2];  // [B,H]
  const float* c0  = (const float*)d_in[3];  // [B,H]
  const float* W   = (const float*)d_in[4];  // [4H,D]
  const float* U   = (const float*)d_in[5];  // [4H,H]
  const float* b   = (const float*)d_in[6];  // [4H]
  const float* Wd  = (const float*)d_in[7];  // [H,H]
  const float* bd  = (const float*)d_in[8];  // [H]
  float* out = (float*)d_out;                // outs [T,B,H] | h [B,H] | c [B,H]

  // Workspace carve-out (~66 MB: dominated by f16 copy of x)
  char* ws = (char*)d_ws;
  size_t off = 0;
  auto carve = [&](size_t bytes) -> void* {
    void* p = ws + off;
    off += (bytes + 255) & ~(size_t)255;
    return p;
  };
  _Float16* xh  = (_Float16*)carve((size_t)T_DIM * B_DIM * D_DIM * sizeof(_Float16));
  _Float16* Wh  = (_Float16*)carve((size_t)4 * H_DIM * D_DIM * sizeof(_Float16));
  _Float16* Uh  = (_Float16*)carve((size_t)4 * H_DIM * H_DIM * sizeof(_Float16));
  _Float16* Wdh = (_Float16*)carve((size_t)H_DIM * H_DIM * sizeof(_Float16));
  _Float16* hbuf[2], * cbufh[2];
  float* cbuff[2];
  for (int i = 0; i < 2; ++i) {
    hbuf[i]  = (_Float16*)carve((size_t)B_DIM * H_DIM * sizeof(_Float16));
    cbufh[i] = (_Float16*)carve((size_t)B_DIM * H_DIM * sizeof(_Float16));
    cbuff[i] = (float*)carve((size_t)B_DIM * H_DIM * sizeof(float));
  }

  const int BH = B_DIM * H_DIM;
  const int thr = 256;

  // One-time conversions (deterministic per call).
  {
    const size_t nx = (size_t)T_DIM * B_DIM * D_DIM;
    cvt_f32_to_h_kernel<<<(int)((nx + thr - 1) / thr), thr, 0, stream>>>(x, xh, (int)nx);
  }
  cvt_f32_to_h_kernel<<<(4 * H_DIM * D_DIM + thr - 1) / thr, thr, 0, stream>>>(W, Wh, 4 * H_DIM * D_DIM);
  cvt_f32_to_h_kernel<<<(4 * H_DIM * H_DIM + thr - 1) / thr, thr, 0, stream>>>(U, Uh, 4 * H_DIM * H_DIM);
  cvt_f32_to_h_kernel<<<(H_DIM * H_DIM + thr - 1) / thr, thr, 0, stream>>>(Wd, Wdh, H_DIM * H_DIM);
  cvt_f32_to_h_kernel<<<(BH + thr - 1) / thr, thr, 0, stream>>>(h0, hbuf[0], BH);
  cvt_f32_to_h_kernel<<<(BH + thr - 1) / thr, thr, 0, stream>>>(c0, cbufh[0], BH);
  copy_f32_kernel<<<(BH + thr - 1) / thr, thr, 0, stream>>>(c0, cbuff[0], BH);

  float* hfin = out + (size_t)T_DIM * BH;
  float* cfin = hfin + BH;

  // Sequential recurrence: one launch per step, ping-pong state buffers.
  for (int t = 0; t < T_DIM; ++t) {
    const int cur = t & 1;
    const int nxt = cur ^ 1;
    const bool last = (t == T_DIM - 1);
    tlstm_step<<<(B_DIM / 16) * (H_DIM / 16), 32, 0, stream>>>(
        xh + (size_t)t * B_DIM * D_DIM, dts, t,
        Wh, Uh, Wdh, b, bd,
        hbuf[cur], cbufh[cur], cbuff[cur],
        hbuf[nxt], cbufh[nxt], cbuff[nxt],
        out,
        last ? hfin : nullptr,
        last ? cfin : nullptr);
  }
}
